// CausalSelfAttention_15668040696062
// MI455X (gfx1250) — matrix operand order repaired and checked by reading, not compiled, not run on hardware
//
#include <hip/hip_runtime.h>
#include <hip/hip_bf16.h>
#include <stdint.h>

typedef _Float16 f16;
typedef __attribute__((ext_vector_type(8)))  _Float16 v8h;
typedef __attribute__((ext_vector_type(16))) _Float16 v16h;
typedef __attribute__((ext_vector_type(8)))  float    v8f;

#define NEG_INF (-__builtin_inff())

// ---------------------------------------------------------------------------
// Fragment loaders matching CDNA5 WMMA VGPR layouts (wave32).
// A-matrix 16x32 f16: lane L -> row m = L&15, half hs = L>>4.
//   elements 0..7  hold K = hs*8 + 0..7 ; elements 8..15 hold K = 16 + hs*8 + 0..7
// ---------------------------------------------------------------------------
__device__ inline v16h load_a_frag(const f16* __restrict__ base, int ld, int lane) {
    const int m  = lane & 15;
    const int hs = (lane >> 4) & 1;
    const f16* p = base + (size_t)m * (size_t)ld + hs * 8;
    v8h lo = *reinterpret_cast<const v8h*>(p);
    v8h hi = *reinterpret_cast<const v8h*>(p + 16);
    v16h r;
#pragma unroll
    for (int i = 0; i < 8; ++i) { r[i] = lo[i]; r[i + 8] = hi[i]; }
    return r;
}

// B-matrix 32x16 f16 (stored as rows of K-contiguous data, one row per column n):
// lane L -> column n = L&15, half hs = L>>4; elements 0..7 hold K = hs*8 + 0..7 ; elements 8..15 hold K = 16 + hs*8 + 0..7
__device__ inline v16h load_b_frag(const f16* __restrict__ base, int ld, int lane) {
    const int n  = lane & 15;
    const int hs = (lane >> 4) & 1;
    const f16* p = base + (size_t)n * (size_t)ld + hs * 8;
    v8h lo = *reinterpret_cast<const v8h*>(p);
    v8h hi = *reinterpret_cast<const v8h*>(p + 16);
    v16h r;
#pragma unroll
    for (int i = 0; i < 8; ++i) { r[i] = lo[i]; r[i + 8] = hi[i]; }
    return r;
}

__device__ inline v8f wmma_f16(v16h a, v16h b, v8f c) {
    return __builtin_amdgcn_wmma_f32_16x16x32_f16(
        /*neg_a=*/false, a, /*neg_b=*/false, b,
        /*c_mod=*/(short)0, c, /*reuse_a=*/false, /*reuse_b=*/false);
}

// Async global->LDS copy, 16 bytes per lane (CDNA5, tracked by ASYNCcnt).
__device__ inline void async_load_b128(uint32_t lds_off, const f16* gptr) {
    asm volatile("global_load_async_to_lds_b128 %0, %1, off"
                 :: "v"(lds_off), "v"(gptr) : "memory");
}
__device__ inline void wait_asynccnt0() {
    asm volatile("s_wait_asynccnt 0x0" ::: "memory");
}

// ---------------------------------------------------------------------------
// f32 -> f16 conversion
// ---------------------------------------------------------------------------
__global__ void cvt_f32_f16(const float* __restrict__ src, f16* __restrict__ dst, size_t n) {
    size_t i = (size_t)blockIdx.x * blockDim.x + threadIdx.x;
    size_t stride = (size_t)gridDim.x * blockDim.x;
    for (; i < n; i += stride) dst[i] = (f16)src[i];
}

// ---------------------------------------------------------------------------
// WMMA GEMM: C[M,N] (f32) = A[M,K] (f16 row-major) x B[N,K]^T (f16 row-major).
// Block = 8 waves = 256x64 block tile; wave tile 32x64.
// B panel (64 cols x 32 K = 4KB) staged in LDS with async copies, dbl-buffered.
// ---------------------------------------------------------------------------
__global__ void gemm_wmma_f16(const f16* __restrict__ A, const f16* __restrict__ B,
                              float* __restrict__ C, int M, int N, int K) {
    __shared__ f16 bbuf[2][64][32];

    const int lane = threadIdx.x & 31;
    const int w    = threadIdx.x >> 5;
    const int nblocks = N >> 6;
    const int mblock  = blockIdx.x / nblocks;
    const int nblock  = blockIdx.x % nblocks;
    const size_t m0 = (size_t)mblock * 256 + (size_t)w * 32;
    const size_t n0 = (size_t)nblock * 64;

    // this wave's slice of the B panel: 512B = one b128 per lane
    const int li    = (w << 5) | lane;       // 0..255
    const int brow  = li >> 2;               // 0..63
    const int bcol  = (li & 3) * 8;          // halves
    const f16* bsrc0 = B + (n0 + brow) * (size_t)K + bcol;

    // prologue: stage k0 = 0
    async_load_b128((uint32_t)(uintptr_t)&bbuf[0][brow][bcol], bsrc0);
    wait_asynccnt0();
    __syncthreads();

    v8f acc[2][4] = {};
    int cur = 0;
    for (int k0 = 0; k0 < K; k0 += 32) {
        // stage next K panel while computing this one
        if (k0 + 32 < K)
            async_load_b128((uint32_t)(uintptr_t)&bbuf[cur ^ 1][brow][bcol],
                            bsrc0 + k0 + 32);

        __builtin_prefetch(A + m0 * (size_t)K + k0 + 256, 0, 3);
        v16h a0 = load_a_frag(A + m0 * (size_t)K + k0,        K, lane);
        v16h a1 = load_a_frag(A + (m0 + 16) * (size_t)K + k0, K, lane);

        // hoist all four B fragments (one DS clause), then 8 back-to-back WMMAs
        v16h b0 = load_b_frag(&bbuf[cur][0][0],  32, lane);
        v16h b1 = load_b_frag(&bbuf[cur][16][0], 32, lane);
        v16h b2 = load_b_frag(&bbuf[cur][32][0], 32, lane);
        v16h b3 = load_b_frag(&bbuf[cur][48][0], 32, lane);

        acc[0][0] = wmma_f16(a0, b0, acc[0][0]);
        acc[1][0] = wmma_f16(a1, b0, acc[1][0]);
        acc[0][1] = wmma_f16(a0, b1, acc[0][1]);
        acc[1][1] = wmma_f16(a1, b1, acc[1][1]);
        acc[0][2] = wmma_f16(a0, b2, acc[0][2]);
        acc[1][2] = wmma_f16(a1, b2, acc[1][2]);
        acc[0][3] = wmma_f16(a0, b3, acc[0][3]);
        acc[1][3] = wmma_f16(a1, b3, acc[1][3]);

        wait_asynccnt0();
        __syncthreads();
        cur ^= 1;
    }

    const int hs = (lane >> 4) & 1;
    const int nn = lane & 15;
#pragma unroll
    for (int mi = 0; mi < 2; ++mi)
#pragma unroll
        for (int j = 0; j < 4; ++j)
#pragma unroll
            for (int r = 0; r < 8; ++r)
                C[(m0 + mi * 16 + r + hs * 8) * (size_t)N + n0 + j * 16 + nn] =
                    acc[mi][j][r];
}

// ---------------------------------------------------------------------------
// QKV epilogue: v = l0*v + l1*ve, rms_norm(q,k), rotary(q,k).
// One 32-lane wave per (b,t,h); lane d handles dims d and d+32.
// Writes qr,kr as [b,h,t,d] f16 and v transposed as [b,h,d,t] f16.
// ---------------------------------------------------------------------------
__global__ void qkv_epilogue(const float* __restrict__ qkv, const float* __restrict__ ve,
                             const float* __restrict__ lambdas,
                             f16* __restrict__ qr, f16* __restrict__ kr, f16* __restrict__ vt,
                             int T, int H, int D) {
    const int lane = threadIdx.x & 31;
    const int item = blockIdx.x * (blockDim.x >> 5) + (threadIdx.x >> 5); // (b*T+t)*H + h
    const int h  = item % H;
    const int bt = item / H;
    const int t  = bt % T;
    const int b  = bt / T;
    const int C  = H * D;

    const size_t rowbase = (size_t)bt * 3 * C + h * D;
    float q1 = qkv[rowbase + lane],         q2 = qkv[rowbase + lane + 32];
    float k1 = qkv[rowbase + C + lane],     k2 = qkv[rowbase + C + lane + 32];
    float v1 = qkv[rowbase + 2 * C + lane], v2 = qkv[rowbase + 2 * C + lane + 32];
    const size_t vebase = (size_t)bt * C + h * D;
    float e1 = ve[vebase + lane], e2 = ve[vebase + lane + 32];
    float l0 = lambdas[0], l1 = lambdas[1];
    v1 = l0 * v1 + l1 * e1;
    v2 = l0 * v2 + l1 * e2;

    // rms_norm over 64 dims (full-wave reduction)
    float ssq = q1 * q1 + q2 * q2;
    float ssk = k1 * k1 + k2 * k2;
#pragma unroll
    for (int m = 16; m >= 1; m >>= 1) {
        ssq += __shfl_xor(ssq, m, 32);
        ssk += __shfl_xor(ssk, m, 32);
    }
    float qs = rsqrtf(ssq * (1.0f / 64.0f) + 1e-6f);
    float ks = rsqrtf(ssk * (1.0f / 64.0f) + 1e-6f);
    q1 *= qs; q2 *= qs; k1 *= ks; k2 *= ks;

    // rotary: nfreq=16 real freqs then 16 zeros; angular[i] = (1/1024)^(i/15)
    float ang = (lane < 16) ? exp2f(-10.0f * (float)lane * (1.0f / 15.0f)) : 0.0f;
    float th = (float)t * ang;
    float c = cosf(th), s = sinf(th);
    float qy1 =  q1 * c + q2 * s, qy2 = -q1 * s + q2 * c;
    float ky1 =  k1 * c + k2 * s, ky2 = -k1 * s + k2 * c;

    const size_t qkb = (((size_t)(b * H + h)) * T + t) * (size_t)D;
    qr[qkb + lane]      = (f16)qy1;
    qr[qkb + lane + 32] = (f16)qy2;
    kr[qkb + lane]      = (f16)ky1;
    kr[qkb + lane + 32] = (f16)ky2;
    const size_t vtb = ((size_t)(b * H + h)) * (size_t)D * (size_t)T + t;
    vt[vtb + (size_t)lane * T]        = (f16)v1;
    vt[vtb + (size_t)(lane + 32) * T] = (f16)v2;
}

// ---------------------------------------------------------------------------
// Flash attention (causal, online softmax). One wave per (b,h, 16-row q tile).
// S = Q K^T * 0.12 ; P = softmax ; O = P V. All matmuls via v_wmma f16->f32.
// ---------------------------------------------------------------------------
__global__ void flash_attn(const f16* __restrict__ qr, const f16* __restrict__ kr,
                           const f16* __restrict__ vt, f16* __restrict__ y,
                           int T, int H, int D) {
    const int lane  = threadIdx.x & 31;
    const int wslot = threadIdx.x >> 5;
    const int wave  = blockIdx.x * (blockDim.x >> 5) + wslot;
    const int QT = T >> 4;
    const int qt = wave % QT;
    const int bh = wave / QT;
    const int h  = bh % H;
    const int b  = bh / H;
    const int tq0 = qt << 4;
    const int C = H * D;

    __shared__ f16 pbuf[4][16][32];
    f16 (*pb)[32] = pbuf[wslot];

    const f16* qbase = qr + (((size_t)bh) * T + tq0) * (size_t)D;
    const f16* kbase = kr + ((size_t)bh) * (size_t)T * (size_t)D;
    const f16* vbase = vt + ((size_t)bh) * (size_t)D * (size_t)T;

    const v16h qa0 = load_a_frag(qbase,      D, lane); // K dims 0..31
    const v16h qa1 = load_a_frag(qbase + 32, D, lane); // K dims 32..63

    v8f o0 = {}, o1 = {}, o2 = {}, o3 = {};
    float mrow[8], lrow[8];
#pragma unroll
    for (int r = 0; r < 8; ++r) { mrow[r] = NEG_INF; lrow[r] = 0.0f; }

    const int hs = (lane >> 4) & 1;
    const int nn = lane & 15;

    for (int j0 = 0; j0 <= tq0 + 15; j0 += 32) {
        // ---- scores: two 16x16 tiles over 32 keys, K-dim = 64 ----
        v16h b0a = load_b_frag(kbase + (size_t)j0 * D,              D, lane);
        v16h b0b = load_b_frag(kbase + (size_t)j0 * D + 32,        D, lane);
        v16h b1a = load_b_frag(kbase + (size_t)(j0 + 16) * D,      D, lane);
        v16h b1b = load_b_frag(kbase + (size_t)(j0 + 16) * D + 32, D, lane);
        // prefetch V fragments early so they overlap the softmax VALU work
        v16h vb0 = load_b_frag(vbase + (size_t)j0,          T, lane);
        v16h vb1 = load_b_frag(vbase + 16 * (size_t)T + j0, T, lane);
        v16h vb2 = load_b_frag(vbase + 32 * (size_t)T + j0, T, lane);
        v16h vb3 = load_b_frag(vbase + 48 * (size_t)T + j0, T, lane);

        v8f s0 = {}, s1 = {};
        s0 = wmma_f16(qa0, b0a, s0);
        s0 = wmma_f16(qa1, b0b, s0);
        s1 = wmma_f16(qa0, b1a, s1);
        s1 = wmma_f16(qa1, b1b, s1);

        const bool need_mask = (j0 + 31 > tq0);
#pragma unroll
        for (int r = 0; r < 8; ++r) {
            const int trow = tq0 + r + hs * 8;
            float sv0 = s0[r] * 0.12f;
            float sv1 = s1[r] * 0.12f;
            if (need_mask) {
                if (j0 + nn > trow)      sv0 = NEG_INF;
                if (j0 + 16 + nn > trow) sv1 = NEG_INF;
            }
            float cm = fmaxf(sv0, sv1);
#pragma unroll
            for (int m = 1; m < 16; m <<= 1) cm = fmaxf(cm, __shfl_xor(cm, m, 32));
            float mnew  = fmaxf(mrow[r], cm);
            float alpha = __expf(mrow[r] - mnew);
            float p0 = __expf(sv0 - mnew);
            float p1 = __expf(sv1 - mnew);
            float rs = p0 + p1;
#pragma unroll
            for (int m = 1; m < 16; m <<= 1) rs += __shfl_xor(rs, m, 32);
            lrow[r] = lrow[r] * alpha + rs;
            mrow[r] = mnew;
            o0[r] *= alpha; o1[r] *= alpha; o2[r] *= alpha; o3[r] *= alpha;
            const int row = r + hs * 8;
            pb[row][nn]      = (f16)p0;
            pb[row][16 + nn] = (f16)p1;
        }
        // same-wave LDS RAW: make the DS stores land before A-fragment reads
        asm volatile("s_wait_dscnt 0" ::: "memory");
        v16h pa = load_a_frag(&pb[0][0], 32, lane);

        // ---- O += P * V : 4 output d-tiles, K-dim = 32 kv ----
        o0 = wmma_f16(pa, vb0, o0);
        o1 = wmma_f16(pa, vb1, o1);
        o2 = wmma_f16(pa, vb2, o2);
        o3 = wmma_f16(pa, vb3, o3);
    }

    // normalize and store y[b,t,h,d] as f16 (A matrix of the c_proj GEMM)
#pragma unroll
    for (int r = 0; r < 8; ++r) {
        float inv = 1.0f / lrow[r];
        size_t ob = ((size_t)b * T + tq0 + r + hs * 8) * (size_t)C + h * D + nn;
        y[ob + 0]  = (f16)(o0[r] * inv);
        y[ob + 16] = (f16)(o1[r] * inv);
        y[ob + 32] = (f16)(o2[r] * inv);
        y[ob + 48] = (f16)(o3[r] * inv);
    }
}

// ---------------------------------------------------------------------------
extern "C" void kernel_launch(void* const* d_in, const int* in_sizes, int n_in,
                              void* d_out, int out_size, void* d_ws, size_t ws_size,
                              hipStream_t stream) {
    const float* x       = (const float*)d_in[0];
    const float* ve      = (const float*)d_in[1];
    const float* qkv_w   = (const float*)d_in[2];
    const float* lambdas = (const float*)d_in[3];
    const float* c_proj  = (const float*)d_in[4];
    float* out = (float*)d_out;

    const int Bb = 4, T = 2048, C = 1024, H = 16, D = 64;
    const size_t M = (size_t)Bb * T; // 8192

    char* ws = (char*)d_ws;
    size_t off = 0;
    f16*   xh  = (f16*)(ws + off);   off += M * C * 2;               // x in f16
    f16*   wh  = (f16*)(ws + off);   off += (size_t)3 * C * C * 2;   // qkv_w f16
    f16*   ph  = (f16*)(ws + off);   off += (size_t)C * C * 2;       // c_proj f16
    float* qkv = (float*)(ws + off); off += M * 3 * C * 4;           // qkv f32
    f16*   qr  = (f16*)(ws + off);   off += M * C * 2;               // q rotated [b,h,t,d]
    f16*   kr  = (f16*)(ws + off);   off += M * C * 2;               // k rotated [b,h,t,d]
    f16*   vt  = (f16*)(ws + off);   off += M * C * 2;               // v mixed   [b,h,d,t]
    f16*   yb  = (f16*)(ws + off);   off += M * C * 2;               // attn out  [b,t,h,d]

    cvt_f32_f16<<<2048, 256, 0, stream>>>(x,      xh, M * C);
    cvt_f32_f16<<<2048, 256, 0, stream>>>(qkv_w,  wh, (size_t)3 * C * C);
    cvt_f32_f16<<<1024, 256, 0, stream>>>(c_proj, ph, (size_t)C * C);

    { // QKV projection: [8192,1024] x [3072,1024]^T -> [8192,3072]
        int blocks = (int)(M / 256) * (3 * C / 64); // 32 * 48
        gemm_wmma_f16<<<blocks, 256, 0, stream>>>(xh, wh, qkv, (int)M, 3 * C, C);
    }

    qkv_epilogue<<<(Bb * T * H) / 4, 128, 0, stream>>>(qkv, ve, lambdas, qr, kr, vt, T, H, D);

    flash_attn<<<(Bb * H * (T / 16)) / 4, 128, 0, stream>>>(qr, kr, vt, yb, T, H, D);

    { // output projection: [8192,1024] x [1024,1024]^T -> [8192,1024]
        int blocks = (int)(M / 256) * (C / 64);    // 32 * 16
        gemm_wmma_f16<<<blocks, 256, 0, stream>>>(yb, ph, out, (int)M, C, C);
    }
}
